// STGNN_Cell_40235253629005
// MI455X (gfx1250) — matrix operand order repaired and checked by reading, not compiled, not run on hardware
//
#include <hip/hip_runtime.h>
#include <hip/hip_bf16.h>

// Problem constants (from reference)
#define BATCH 16384
#define NN    23
#define FEAT  16
#define HID   128
#define G3    (3 * HID)

typedef __attribute__((ext_vector_type(16))) __bf16 v16bf;
typedef __attribute__((ext_vector_type(8)))  __bf16 v8bf;
typedef __attribute__((ext_vector_type(8)))  float  v8f;

// LDS pitches (bf16 elements); all vector ds accesses stay 16B/32B aligned.
#define PX    16    // sX:    32 x 16 (K 16..31 handled as compile-time zeros)
#define PADJ  32    // sAdj:  32 x 32
#define PSUPT 32    // sSupT: [128][32]  (support transposed: [h][m])
#define PROW  136   // sSpat/sH: [32][136] (128 + 8 pad, 272B row = 16B multiple)

__device__ __forceinline__ v16bf make16(v8bf lo, v8bf hi) {
  v16bf r;
#pragma unroll
  for (int i = 0; i < 8; ++i) { r[i] = lo[i]; r[8 + i] = hi[i]; }
  return r;
}

__device__ __forceinline__ v8bf zero8b() {
  v8bf r;
#pragma unroll
  for (int i = 0; i < 8; ++i) r[i] = (__bf16)0.0f;
  return r;
}

__device__ __forceinline__ v8f zero8f() {
  v8f r;
#pragma unroll
  for (int i = 0; i < 8; ++i) r[i] = 0.0f;
  return r;
}

__device__ __forceinline__ float fsig(float x)  { return 1.0f / (1.0f + __expf(-x)); }
__device__ __forceinline__ float ftanhf(float x){ float e = __expf(2.0f * x); return (e - 1.0f) / (e + 1.0f); }

// ---------------------------------------------------------------------------
// Prep: convert weights to bf16 once. W_gcn is stored K-padded [128][32]
// (cols 16..31 zero) so stage-1 B-fragments load with full EXEC, no branches.
// ---------------------------------------------------------------------------
__global__ void stgnn_prep(const float* __restrict__ Wg,
                           const float* __restrict__ Wih,
                           const float* __restrict__ Whh,
                           __bf16* __restrict__ oWg2,   // [128][32] padded
                           __bf16* __restrict__ oWih,
                           __bf16* __restrict__ oWhh) {
  int i = blockIdx.x * blockDim.x + threadIdx.x;
  if (i < HID * 32) {
    int h = i >> 5, f = i & 31;
    oWg2[i] = (f < FEAT) ? (__bf16)Wg[h * FEAT + f] : (__bf16)0.0f;
  }
  if (i < G3 * HID) {
    oWih[i] = (__bf16)Wih[i];
    oWhh[i] = (__bf16)Whh[i];
  }
}

// ---------------------------------------------------------------------------
// Main: one workgroup (8 wave32) per batch element. Each wave owns hidden
// column block j = wave (16 cols) and sweeps both 16-row M-tiles, so every
// B-fragment (weights / staged support) is loaded once and reused.
// ---------------------------------------------------------------------------
__global__ __launch_bounds__(256)
void stgnn_main(const float* __restrict__ x,
                const float* __restrict__ adj,
                const float* __restrict__ hprev,
                const __bf16* __restrict__ Wg2,   // [128][32] bf16, K-padded
                const float*  __restrict__ bg,    // [128]
                const __bf16* __restrict__ Wih,   // [384][128] bf16
                const __bf16* __restrict__ Whh,   // [384][128] bf16
                const float*  __restrict__ bih,   // [384]
                const float*  __restrict__ bhh,   // [384]
                float* __restrict__ out) {
  __shared__ __bf16 sX[32 * PX];
  __shared__ __bf16 sAdj[32 * PADJ];
  __shared__ __bf16 sSupT[HID * PSUPT];
  __shared__ __bf16 sSpat[32 * PROW];
  __shared__ __bf16 sH[32 * PROW];

  const int tid  = threadIdx.x;
  const int lane = tid & 31;
  const int wave = tid >> 5;
  const int b    = blockIdx.x;

  const int colL = lane & 15;   // N-column within 16-wide tile
  const int half = lane >> 4;   // lane half: K-chunk select (A/B), M+8 (C/D)
  const int kbA  = half * 16;   // A-frag K 16-block base
  const int kbB  = half * 16;   // B-frag K 16-block base
  const int j    = wave;        // this wave's hidden column block

  // Warm gate weights in WGP$/GL2 (global_prefetch_b8).
  __builtin_prefetch(Wih + (size_t)tid * 192, 0, 1);
  __builtin_prefetch(Whh + (size_t)tid * 192, 0, 1);

  // Zero padded staging regions.
  for (int i = tid; i < 32 * PX;   i += 256) sX[i]   = (__bf16)0.0f;
  for (int i = tid; i < 32 * PADJ; i += 256) sAdj[i] = (__bf16)0.0f;
  for (int i = tid; i < 32 * PROW; i += 256) sH[i]   = (__bf16)0.0f;
  __syncthreads();

  // Stage activations as bf16.
  const float* xb = x     + (size_t)b * NN * FEAT;
  const float* ab = adj   + (size_t)b * NN * NN;
  const float* hb = hprev + (size_t)b * NN * HID;
  for (int i = tid; i < NN * FEAT; i += 256)
    sX[(i / FEAT) * PX + (i % FEAT)] = (__bf16)xb[i];
  for (int i = tid; i < NN * NN; i += 256)
    sAdj[(i / NN) * PADJ + (i % NN)] = (__bf16)ab[i];
  for (int i = tid; i < NN * HID; i += 256)
    sH[(i / HID) * PROW + (i % HID)] = (__bf16)hb[i];
  __syncthreads();

  // ---------------- Stage 1: support = x * Wgcn^T + bg  (K=16, padded to 32)
  {
    // B-fragment from padded weights: all lanes load (K 16..31 rows are zero).
    const v16bf bfr = *(const v16bf*)(Wg2 + (size_t)(j * 16 + colL) * 32 + kbB);
    const float bias = bg[j * 16 + colL];
#pragma unroll
    for (int m = 0; m < 2; ++m) {
      const int rowA = m * 16 + colL;
      v16bf a = half ? make16(zero8b(), zero8b())
                     : make16(*(const v8bf*)&sX[rowA * PX], *(const v8bf*)&sX[rowA * PX + 8]);
      v8f c = zero8f();
      c = __builtin_amdgcn_wmma_f32_16x16x32_bf16(false, a, false, bfr,
                                                  (short)0, c, false, false);
#pragma unroll
      for (int v = 0; v < 8; ++v) {
        const int row = m * 16 + half * 8 + v;
        sSupT[(j * 16 + colL) * PSUPT + row] = (__bf16)(c[v] + bias);
      }
    }
  }
  __syncthreads();

  // ---------------- Stage 2: spatial = relu(adj * support)  (K=32, padded)
  {
    // B[k][col] = support[m=k][h=j*16+col] -> contiguous in sSupT[h][m].
    const v16bf bfr = *(const v16bf*)&sSupT[(j * 16 + colL) * PSUPT + kbB];
#pragma unroll
    for (int m = 0; m < 2; ++m) {
      const int rowA = m * 16 + colL;
      v16bf a = make16(*(const v8bf*)&sAdj[rowA * PADJ + kbA],
                       *(const v8bf*)&sAdj[rowA * PADJ + kbA + 8]);
      v8f c = zero8f();
      c = __builtin_amdgcn_wmma_f32_16x16x32_bf16(false, a, false, bfr,
                                                  (short)0, c, false, false);
#pragma unroll
      for (int v = 0; v < 8; ++v) {
        const int row = m * 16 + half * 8 + v;
        sSpat[row * PROW + (j * 16 + colL)] = (__bf16)fmaxf(c[v], 0.0f);
      }
    }
  }
  __syncthreads();

  // ---------------- Stage 3: fused GRU gates (K=128): each B-fragment loaded
  // once, used for both M-tiles. 4 K-steps x 3 gates x (ih+hh) x 2 M = 48 WMMA.
  {
    v8f ci[2][3], ch[2][3];
#pragma unroll
    for (int m = 0; m < 2; ++m)
#pragma unroll
      for (int g = 0; g < 3; ++g) { ci[m][g] = zero8f(); ch[m][g] = zero8f(); }

#pragma unroll
    for (int kb = 0; kb < 4; ++kb) {
      const int ka = kb * 32 + kbA;
      const v16bf aS0 = make16(*(const v8bf*)&sSpat[(0 * 16 + colL) * PROW + ka],
                               *(const v8bf*)&sSpat[(0 * 16 + colL) * PROW + ka + 8]);
      const v16bf aS1 = make16(*(const v8bf*)&sSpat[(1 * 16 + colL) * PROW + ka],
                               *(const v8bf*)&sSpat[(1 * 16 + colL) * PROW + ka + 8]);
      const v16bf aH0 = make16(*(const v8bf*)&sH[(0 * 16 + colL) * PROW + ka],
                               *(const v8bf*)&sH[(0 * 16 + colL) * PROW + ka + 8]);
      const v16bf aH1 = make16(*(const v8bf*)&sH[(1 * 16 + colL) * PROW + ka],
                               *(const v8bf*)&sH[(1 * 16 + colL) * PROW + ka + 8]);
      const int kB = kb * 32 + kbB;
#pragma unroll
      for (int g = 0; g < 3; ++g) {
        const size_t wrow = (size_t)(g * HID + j * 16 + colL) * HID + kB;
        const v16bf bi = *(const v16bf*)(Wih + wrow);
        ci[0][g] = __builtin_amdgcn_wmma_f32_16x16x32_bf16(false, aS0, false, bi,
                                                           (short)0, ci[0][g], false, false);
        ci[1][g] = __builtin_amdgcn_wmma_f32_16x16x32_bf16(false, aS1, false, bi,
                                                           (short)0, ci[1][g], false, false);
        const v16bf bh = *(const v16bf*)(Whh + wrow);
        ch[0][g] = __builtin_amdgcn_wmma_f32_16x16x32_bf16(false, aH0, false, bh,
                                                           (short)0, ch[0][g], false, false);
        ch[1][g] = __builtin_amdgcn_wmma_f32_16x16x32_bf16(false, aH1, false, bh,
                                                           (short)0, ch[1][g], false, false);
      }
    }

    // GRU epilogue + guarded store (rows 23..31 are padding).
    const int hcol = j * 16 + colL;
    const float bir = bih[hcol], biz = bih[HID + hcol], bin = bih[2 * HID + hcol];
    const float bhr = bhh[hcol], bhz = bhh[HID + hcol], bhn = bhh[2 * HID + hcol];

#pragma unroll
    for (int m = 0; m < 2; ++m) {
#pragma unroll
      for (int v = 0; v < 8; ++v) {
        const int row = m * 16 + half * 8 + v;
        if (row < NN) {
          const float r  = fsig((ci[m][0][v] + bir) + (ch[m][0][v] + bhr));
          const float z  = fsig((ci[m][1][v] + biz) + (ch[m][1][v] + bhz));
          const float hn = ch[m][2][v] + bhn;
          const float n  = ftanhf((ci[m][2][v] + bin) + r * hn);
          const float hp = hb[row * HID + hcol];
          out[(size_t)b * NN * HID + (size_t)row * HID + hcol] =
              (1.0f - z) * n + z * hp;
        }
      }
    }
  }
}

// ---------------------------------------------------------------------------
extern "C" void kernel_launch(void* const* d_in, const int* in_sizes, int n_in,
                              void* d_out, int out_size, void* d_ws, size_t ws_size,
                              hipStream_t stream) {
  const float* x     = (const float*)d_in[0];
  const float* adj   = (const float*)d_in[1];
  const float* hprev = (const float*)d_in[2];
  const float* Wg    = (const float*)d_in[3];
  const float* bg    = (const float*)d_in[4];
  const float* Wih   = (const float*)d_in[5];
  const float* Whh   = (const float*)d_in[6];
  const float* bih   = (const float*)d_in[7];
  const float* bhh   = (const float*)d_in[8];
  float* out = (float*)d_out;

  // Workspace layout (bf16 weight copies), 256B-aligned segments.
  char* ws = (char*)d_ws;
  __bf16* wg_bf  = (__bf16*)ws;                         // 128*32*2  = 8192 B
  __bf16* wih_bf = (__bf16*)(ws + 8192);                // 49152*2   = 98304 B
  __bf16* whh_bf = (__bf16*)(ws + 8192 + 98304);        // 49152*2   = 98304 B

  stgnn_prep<<<(G3 * HID + 255) / 256, 256, 0, stream>>>(Wg, Wih, Whh,
                                                         wg_bf, wih_bf, whh_bf);
  stgnn_main<<<BATCH, 256, 0, stream>>>(x, adj, hprev, wg_bf, bg,
                                        wih_bf, whh_bf, bih, bhh, out);
}
